// GrokkingSNN_249108103373
// MI455X (gfx1250) — matrix operand order repaired and checked by reading, not compiled, not run on hardware
//
#include <hip/hip_runtime.h>
#include <hip/hip_bf16.h>
#include <stdint.h>

// ---- problem constants (match reference) ----
#define PSZ       2048
#define HIDDEN    512
#define TWOH      1024
#define BATCH     8192
#define NUM_STEPS 15

typedef __attribute__((ext_vector_type(16))) _Float16 v16h;
typedef __attribute__((ext_vector_type(8)))  _Float16 v8h;
typedef __attribute__((ext_vector_type(8)))  float    v8f;

// async global->LDS staging (GVS mode: lds-addr VGPR, 32-bit voffset VGPR, SGPR64 base)
#define ASYNC_LDS_B128(ldsaddr, voff, base)                                \
  asm volatile("global_load_async_to_lds_b128 %0, %1, %2"                  \
               :: "v"(ldsaddr), "v"(voff), "s"(base) : "memory")
#define WAIT_ASYNC0() asm volatile("s_wait_asynccnt 0x0" ::: "memory")

// -------------------------------------------------------------------------
// Kernel 1: convert/transpose weights to f16, WMMA-B friendly [N][K] layout.
//   W1 [TWOH][HIDDEN] f32  -> W1T [HIDDEN][TWOH] f16
//   W2 [HIDDEN][PSZ]  f32  -> W2T [PSZ][HIDDEN]  f16
// -------------------------------------------------------------------------
__global__ __launch_bounds__(256)
void snn_convert_weights(const float* __restrict__ W1,
                         const float* __restrict__ W2,
                         _Float16* __restrict__ W1T,
                         _Float16* __restrict__ W2T) {
  const int i = blockIdx.x * 256 + threadIdx.x;
  const int n1 = HIDDEN * TWOH;            // 524288
  const int n2 = PSZ * HIDDEN;             // 1048576
  if (i < n1) {
    const int n = i / TWOH;
    const int k = i - n * TWOH;
    W1T[i] = (_Float16)W1[(size_t)k * HIDDEN + n];
  } else if (i < n1 + n2) {
    const int j = i - n1;
    const int n = j / HIDDEN;
    const int k = j - n * HIDDEN;
    W2T[j] = (_Float16)W2[(size_t)k * PSZ + n];
  }
}

// -------------------------------------------------------------------------
// Kernel 2: fused  cur1 = gather(embed,x) @ W1 + b1  + 15-step LIF recurrence.
// Emits S = sum_t beta2^(15-t)*spk1_t as f16 [BATCH][HIDDEN].
// One wave32 computes a 32(M) x 64(N) tile: B fragments reused for 2 M-tiles.
// -------------------------------------------------------------------------
__global__ __launch_bounds__(256)
void snn_layer1(const int*   __restrict__ x,       // [BATCH][2]
                const float* __restrict__ embed,   // [PSZ][HIDDEN]
                const _Float16* __restrict__ W1T,  // [HIDDEN][TWOH]
                const float* __restrict__ b1,      // [HIDDEN]
                const float* __restrict__ beta1p,
                const float* __restrict__ thr1p,
                const float* __restrict__ beta2p,
                _Float16* __restrict__ S)          // [BATCH][HIDDEN]
{
  const int lane = threadIdx.x & 31;
  const int wid  = blockIdx.x * 8 + (threadIdx.x >> 5);
  const int mt   = wid >> 3;               // 256 tiles of 32 rows
  const int nt   = wid & 7;                // 8 N tiles of 64
  const int m0   = mt * 32;
  const int n0   = nt * 64;
  const int lm   = lane & 15;
  const bool hi  = lane >= 16;

  // gathered embedding row pointers: A-tile0 row (m0+lm), A-tile1 row (m0+16+lm)
  const int mg0 = m0 + lm, mg1 = m0 + 16 + lm;
  const float* r0a = embed + (size_t)x[2 * mg0 + 0] * HIDDEN;
  const float* r0b = embed + (size_t)x[2 * mg0 + 1] * HIDDEN;
  const float* r1a = embed + (size_t)x[2 * mg1 + 0] * HIDDEN;
  const float* r1b = embed + (size_t)x[2 * mg1 + 1] * HIDDEN;

  v8f acc[2][4];
#pragma unroll
  for (int m = 0; m < 2; ++m)
#pragma unroll
    for (int j = 0; j < 4; ++j)
#pragma unroll
      for (int e = 0; e < 8; ++e) acc[m][j][e] = 0.0f;

  for (int k0 = 0; k0 < TWOH; k0 += 32) {
    // A fragments (contiguous K, matching B): lane<16 K {k0..+15},
    // lane>=16 K {k0+16..+31}; convert gathered f32 -> f16 in regs.
    const int ks0 = k0 + (hi ? 16 : 0);
    const int ks1 = ks0 + 8;
    const v8f f00 = *(const v8f*)((ks0 < HIDDEN ? r0a : r0b) + (ks0 & (HIDDEN - 1)));
    const v8f f01 = *(const v8f*)((ks1 < HIDDEN ? r0a : r0b) + (ks1 & (HIDDEN - 1)));
    const v8f f10 = *(const v8f*)((ks0 < HIDDEN ? r1a : r1b) + (ks0 & (HIDDEN - 1)));
    const v8f f11 = *(const v8f*)((ks1 < HIDDEN ? r1a : r1b) + (ks1 & (HIDDEN - 1)));
    v16h a0, a1;
#pragma unroll
    for (int e = 0; e < 8; ++e) {
      a0[e] = (_Float16)f00[e]; a0[8 + e] = (_Float16)f01[e];
      a1[e] = (_Float16)f10[e]; a1[8 + e] = (_Float16)f11[e];
    }
    const int kb = k0 + (hi ? 16 : 0);
#pragma unroll
    for (int j = 0; j < 4; ++j) {
      const v16h b = *(const v16h*)(W1T + (size_t)(n0 + j * 16 + lm) * TWOH + kb);
      acc[0][j] = __builtin_amdgcn_wmma_f32_16x16x32_f16(
          false, a0, false, b, (short)0, acc[0][j], false, false);
      acc[1][j] = __builtin_amdgcn_wmma_f32_16x16x32_f16(
          false, a1, false, b, (short)0, acc[1][j], false, false);
    }
  }

  // fused LIF recurrence, fully in registers
  const float beta1 = fminf(fmaxf(beta1p[0], 0.1f), 0.9f);
  const float beta2 = fminf(fmaxf(beta2p[0], 0.1f), 0.9f);
  const float thr1  = fmaxf(thr1p[0], 0.1f);

#pragma unroll
  for (int m = 0; m < 2; ++m)
#pragma unroll
    for (int j = 0; j < 4; ++j) {
      const int n = n0 + j * 16 + lm;
      const float bias = b1[n];
#pragma unroll
      for (int r = 0; r < 8; ++r) {
        const float cur = acc[m][j][r] + bias;
        float mem = 0.0f, s = 0.0f;
#pragma unroll
        for (int t = 0; t < NUM_STEPS; ++t) {
          const float reset = (mem > thr1) ? thr1 : 0.0f;  // subtract-reset, prev mem
          mem = beta1 * mem + cur - reset;
          const float spk = (mem > thr1) ? 1.0f : 0.0f;    // Heaviside(mem - thr1)
          s = beta2 * s + spk;
        }
        const int mrow = m0 + m * 16 + (hi ? 8 + r : r);   // C/D: VGPR r <-> M=r / 8+r
        S[(size_t)mrow * HIDDEN + n] = (_Float16)s;
      }
    }
}

// -------------------------------------------------------------------------
// Kernel 3: mem2 = S @ W2 + c*b2,  c = sum_{t} beta2^(15-t).
// Block = 8 waves = 256(M) x 64(N); wave = 32(M) x 64(N).
// B tile (64N x 32K f16) is staged to LDS with global_load_async_to_lds_b128
// (double-buffered, ASYNCcnt + barrier synchronized) and shared by all waves.
// -------------------------------------------------------------------------
__global__ __launch_bounds__(256)
void snn_layer2(const _Float16* __restrict__ S,    // [BATCH][HIDDEN]
                const _Float16* __restrict__ W2T,  // [PSZ][HIDDEN]
                const float* __restrict__ b2,      // [PSZ]
                const float* __restrict__ beta2p,
                float* __restrict__ out)           // [BATCH][PSZ]
{
  // LDS B tile: 64 rows x 32 halves, pitch 40 halves (80B: bank-staggered,
  // 16B aligned for b128 async stores).  Two buffers, 10 KB total.
  __shared__ _Float16 ldsB[2][64 * 40];

  const int tid  = threadIdx.x;
  const int lane = tid & 31;
  const int wv   = tid >> 5;               // 0..7
  const int bm   = blockIdx.x >> 5;        // 32 M blocks of 256
  const int bn   = blockIdx.x & 31;        // 32 N tiles of 64
  const int m0   = bm * 256 + wv * 32;
  const int n0   = bn * 64;
  const int lm   = lane & 15;
  const bool hi  = lane >= 16;

  // --- async staging setup: thread t copies 16B: row r = t/4, seg = t%4 ---
  const int r   = tid >> 2;
  const int seg = tid & 3;
  const unsigned ldsbase = (unsigned)(uintptr_t)&ldsB[0][0];
  const unsigned dst0 = ldsbase + (unsigned)(r * 80 + seg * 16);
  const unsigned dst1 = dst0 + 64u * 80u;
  unsigned voff = (unsigned)(((n0 + r) * HIDDEN + seg * 8) * 2); // bytes into W2T

  const _Float16* arow0 = S + (size_t)(m0 + lm) * HIDDEN;
  const _Float16* arow1 = S + (size_t)(m0 + 16 + lm) * HIDDEN;

  v8f acc[2][4];
#pragma unroll
  for (int m = 0; m < 2; ++m)
#pragma unroll
    for (int j = 0; j < 4; ++j)
#pragma unroll
      for (int e = 0; e < 8; ++e) acc[m][j][e] = 0.0f;

  // prologue: stage K-block 0 into buffer 0
  ASYNC_LDS_B128(dst0, voff, W2T);

  const int NK = HIDDEN / 32;              // 16 K-steps
  for (int i = 0; i < NK; ++i) {
    WAIT_ASYNC0();                          // our slice of buf[i&1] landed
    __syncthreads();                        // all slices landed; prev reads done
    if (i + 1 < NK) {                       // stage next K-block into other buf
      voff += 64;
      ASYNC_LDS_B128((i & 1) ? dst0 : dst1, voff, W2T);
    }

    const int k0  = i * 32;
    const int ks0 = k0 + (hi ? 16 : 0);
    const int ks1 = ks0 + 8;
    const v8h h00 = *(const v8h*)(arow0 + ks0);
    const v8h h01 = *(const v8h*)(arow0 + ks1);
    const v8h h10 = *(const v8h*)(arow1 + ks0);
    const v8h h11 = *(const v8h*)(arow1 + ks1);
    v16h a0, a1;
#pragma unroll
    for (int e = 0; e < 8; ++e) {
      a0[e] = h00[e]; a0[8 + e] = h01[e];
      a1[e] = h10[e]; a1[8 + e] = h11[e];
    }

    const int kbl = hi ? 16 : 0;
    const _Float16* Bt = &ldsB[i & 1][0];
#pragma unroll
    for (int j = 0; j < 4; ++j) {
      const _Float16* pb = Bt + (j * 16 + lm) * 40 + kbl;
      const v8h b0 = *(const v8h*)pb;        // ds_load_b128
      const v8h b1 = *(const v8h*)(pb + 8);  // ds_load_b128
      v16h b;
#pragma unroll
      for (int e = 0; e < 8; ++e) { b[e] = b0[e]; b[8 + e] = b1[e]; }
      acc[0][j] = __builtin_amdgcn_wmma_f32_16x16x32_f16(
          false, a0, false, b, (short)0, acc[0][j], false, false);
      acc[1][j] = __builtin_amdgcn_wmma_f32_16x16x32_f16(
          false, a1, false, b, (short)0, acc[1][j], false, false);
    }
  }

  // bias: each of the 15 steps adds b2, decayed geometrically by beta2
  const float beta2 = fminf(fmaxf(beta2p[0], 0.1f), 0.9f);
  float c = 0.0f;
#pragma unroll
  for (int t = 0; t < NUM_STEPS; ++t) c = c * beta2 + 1.0f;

#pragma unroll
  for (int m = 0; m < 2; ++m)
#pragma unroll
    for (int j = 0; j < 4; ++j) {
      const int n = n0 + j * 16 + lm;
      const float bias = c * b2[n];
#pragma unroll
      for (int rr = 0; rr < 8; ++rr) {
        const int mrow = m0 + m * 16 + (hi ? 8 + rr : rr);
        out[(size_t)mrow * PSZ + n] = acc[m][j][rr] + bias;
      }
    }
}

// -------------------------------------------------------------------------
// Launch.  Inputs (dict order): 0 x[i32], 1 embed, 2 W1, 3 b1, 4 W2, 5 b2,
// 6 beta1, 7 thr1, 8 beta2, 9 thr2(unused).  Output: mem2 f32 [8192*2048].
// Workspace: W1T f16 @0 (1MB), W2T f16 @1MB (2MB), S f16 @3MB (8MB).
// -------------------------------------------------------------------------
extern "C" void kernel_launch(void* const* d_in, const int* in_sizes, int n_in,
                              void* d_out, int out_size, void* d_ws, size_t ws_size,
                              hipStream_t stream) {
  const int*   x      = (const int*)  d_in[0];
  const float* embed  = (const float*)d_in[1];
  const float* W1     = (const float*)d_in[2];
  const float* b1     = (const float*)d_in[3];
  const float* W2     = (const float*)d_in[4];
  const float* b2     = (const float*)d_in[5];
  const float* beta1p = (const float*)d_in[6];
  const float* thr1p  = (const float*)d_in[7];
  const float* beta2p = (const float*)d_in[8];
  float* out = (float*)d_out;

  char* ws = (char*)d_ws;
  _Float16* W1T = (_Float16*)(ws);
  _Float16* W2T = (_Float16*)(ws + (1u << 20));
  _Float16* Sb  = (_Float16*)(ws + (3u << 20));

  // 1) weight convert/transpose
  {
    const int total = HIDDEN * TWOH + PSZ * HIDDEN;   // 1572864
    snn_convert_weights<<<(total + 255) / 256, 256, 0, stream>>>(W1, W2, W1T, W2T);
  }
  // 2) fused gather + GEMM1 + LIF recurrence -> S
  {
    const int waves = (BATCH / 32) * (HIDDEN / 64);   // 2048
    snn_layer1<<<waves / 8, 256, 0, stream>>>(x, embed, W1T, b1,
                                              beta1p, thr1p, beta2p, Sb);
  }
  // 3) GEMM2 (LDS-staged B, async double-buffered) + bias -> out
  {
    const int blocks = (BATCH / 256) * (PSZ / 64);    // 1024
    snn_layer2<<<blocks, 256, 0, stream>>>(Sb, W2T, b2, beta2p, out);
  }
}
